// OldLayer_2843268350511
// MI455X (gfx1250) — matrix operand order repaired and checked by reading, not compiled, not run on hardware
//
#include <hip/hip_runtime.h>
#include <hip/hip_bf16.h>
#include <math.h>

// ---------------- problem constants ----------------
#define Bc   8
#define Sc   1024
#define Cc   768
#define Hc   12
#define Dc   64
#define Fc   3072
#define Mc   (Bc * Sc)          // 8192 rows

typedef __attribute__((ext_vector_type(16))) _Float16 v16h;
typedef __attribute__((ext_vector_type(8)))  float    v8f;

union Frag16 {                   // 32 bytes: 16 f16 elements == 8 VGPRs
    uint4 q[2];
    v16h  h;
};
union H8 {                       // 16 bytes: 8 f16
    uint4    q;
    _Float16 hh[8];
};

__device__ __forceinline__ v8f wmma16(v16h a, v16h b, v8f c) {
    // D = A(16x32 f16) * B(32x16 f16) + C(16x16 f32)
    return __builtin_amdgcn_wmma_f32_16x16x32_f16(
        /*neg_a=*/false, a, /*neg_b=*/false, b,
        /*c_mod=*/(short)0, c, /*reuse_a=*/false, /*reuse_b=*/false);
}

// CDNA5 async global->LDS copy, 16B per lane; tracked by ASYNCcnt.
__device__ __forceinline__ void async_copy_b128(unsigned lds_off, const void* gaddr) {
    asm volatile("global_load_async_to_lds_b128 %0, %1, off"
                 :: "v"(lds_off), "v"(gaddr) : "memory");
}
__device__ __forceinline__ void wait_async0() {
    asm volatile("s_wait_asynccnt 0x0" ::: "memory");
}
__device__ __forceinline__ unsigned lds_addr32(const void* p) {
    // generic shared pointer: low 32 bits are the wave-relative LDS offset
    return (unsigned)(unsigned long long)p;
}

// ---------------- f32 -> f16 weight conversion ----------------
__global__ __launch_bounds__(256) void cvt_f32_f16(const float* __restrict__ s,
                                                   _Float16* __restrict__ d, int n) {
    int i = blockIdx.x * 256 + threadIdx.x;
    if (i < n) d[i] = (_Float16)s[i];
}

// ---------------- LayerNorm (f32 in, f16 out) ----------------
__global__ __launch_bounds__(256) void ln_f16(const float* __restrict__ x,
                                              const float* __restrict__ g,
                                              const float* __restrict__ bta,
                                              _Float16* __restrict__ o) {
    int row = blockIdx.x;
    const float* xr = x + (size_t)row * Cc;
    float s = 0.f, s2 = 0.f;
    for (int c = threadIdx.x; c < Cc; c += 256) {
        float v = xr[c];
        s += v; s2 += v * v;
    }
    __shared__ float rs[256];
    __shared__ float rq[256];
    int tid = threadIdx.x;
    rs[tid] = s; rq[tid] = s2;
    __syncthreads();
    for (int st = 128; st > 0; st >>= 1) {
        if (tid < st) { rs[tid] += rs[tid + st]; rq[tid] += rq[tid + st]; }
        __syncthreads();
    }
    float mu  = rs[0] * (1.0f / Cc);
    float var = rq[0] * (1.0f / Cc) - mu * mu;
    float inv = rsqrtf(var + 1e-6f);
    _Float16* orow = o + (size_t)row * Cc;
    for (int c = threadIdx.x; c < Cc; c += 256)
        orow[c] = (_Float16)((xr[c] - mu) * inv * g[c] + bta[c]);
}

// ---------------- WMMA GEMM: out = epi(A[M,K]*B[K,N] + bias) ----------------
// Double-buffered LDS staged by GLOBAL_LOAD_ASYNC_TO_LDS_B128 (ASYNCcnt).
// EPI 0: f16 out = acc + bias                      (QKV projections)
// EPI 1: f32 out = (acc+bias)*scale[n] + resid     (O-proj / FC2 + layer-scale + residual)
// EPI 2: f16 out = gelu_exact(acc + bias)          (FC1)
template <int EPI>
__global__ __launch_bounds__(256) void gemm_wmma_f16(
        const _Float16* __restrict__ A, const _Float16* __restrict__ Bw,
        const float* __restrict__ bias, const float* __restrict__ scale,
        const float* __restrict__ resid, void* __restrict__ outv,
        int Ndim, int Kdim) {
    __shared__ __align__(16) _Float16 As[2][64][40];    // 64 rows x 32 K (+pad)
    __shared__ __align__(16) _Float16 Bs[2][32][136];   // 32 K x 128 N (+pad)

    const int m0 = blockIdx.y * 64;
    const int n0 = blockIdx.x * 128;
    const int tid  = threadIdx.x;
    const int w    = tid >> 5;
    const int lane = tid & 31;
    const int lr   = lane & 15;
    const int hi   = lane >> 4;          // lane half
    const int wm   = w >> 2;             // 0..1  (rows)
    const int wn   = w & 3;              // 0..3  (cols)

    // per-thread staging coordinates
    const int arow = tid >> 2;           // 0..63
    const int aq   = (tid & 3) * 8;      // K offset (8 halfs = 16B)
    const int brow = tid >> 3;           // 0..31
    const int bc   = (tid & 7) * 16;     // N offset (16 halfs = 32B)
    const _Float16* aptr = A  + (size_t)(m0 + arow) * Kdim + aq;   // + k0
    const _Float16* bptr = Bw + (size_t)brow * Ndim + n0 + bc;     // + k0*Ndim

    v8f acc[2][2];
    for (int i = 0; i < 2; ++i)
        for (int j = 0; j < 2; ++j)
            for (int r = 0; r < 8; ++r) acc[i][j][r] = 0.f;

    // prologue: async-stage k0 = 0 into buffer 0
    async_copy_b128(lds_addr32(&As[0][arow][aq]), aptr);
    async_copy_b128(lds_addr32(&Bs[0][brow][bc]),     bptr);
    async_copy_b128(lds_addr32(&Bs[0][brow][bc + 8]), bptr + 8);

    int it = 0;
    for (int k0 = 0; k0 < Kdim; k0 += 32, ++it) {
        wait_async0();       // this wave's staged tile is in LDS
        __syncthreads();     // everyone's tile is in LDS; prev reads of other buf done
        const int cur = it & 1;

        if (k0 + 32 < Kdim) {  // prefetch next K-step into the other buffer
            const int nxt = cur ^ 1;
            async_copy_b128(lds_addr32(&As[nxt][arow][aq]), aptr + (k0 + 32));
            const _Float16* bp = bptr + (size_t)(k0 + 32) * Ndim;
            async_copy_b128(lds_addr32(&Bs[nxt][brow][bc]),     bp);
            async_copy_b128(lds_addr32(&Bs[nxt][brow][bc + 8]), bp + 8);
        }

        // A fragments (ISA A-layout: lane half selects K sub-run)
        Frag16 af[2];
        for (int mt = 0; mt < 2; ++mt) {
            int row  = wm * 32 + mt * 16 + lr;
            int base = hi * 8;
            af[mt].q[0] = *(const uint4*)(&As[cur][row][base]);        // K = base..base+7
            af[mt].q[1] = *(const uint4*)(&As[cur][row][base + 16]);   // K = base+16..+23
        }
        // B fragments (lane = column, same K order as A: K = hi*8+e, 16+hi*8+(e-8))
        Frag16 bf[2];
        for (int nt = 0; nt < 2; ++nt) {
            int n = wn * 32 + nt * 16 + lr;
            for (int e = 0; e < 16; ++e)
                bf[nt].h[e] = Bs[cur][((e < 8) ? 0 : 16) + hi * 8 + (e & 7)][n];
        }
        for (int mt = 0; mt < 2; ++mt)
            for (int nt = 0; nt < 2; ++nt)
                acc[mt][nt] = wmma16(af[mt].h, bf[nt].h, acc[mt][nt]);
    }

    // epilogue (C/D layout: M = r + 8*hi, N = lane%16)
    for (int mt = 0; mt < 2; ++mt)
        for (int nt = 0; nt < 2; ++nt) {
            int n = n0 + wn * 32 + nt * 16 + lr;
            float bi = bias[n];
            float sc = (EPI == 1) ? scale[n] : 0.f;
            for (int r = 0; r < 8; ++r) {
                int m = m0 + wm * 32 + mt * 16 + r + hi * 8;
                size_t idx = (size_t)m * Ndim + n;
                float v = acc[mt][nt][r] + bi;
                if (EPI == 0) {
                    ((_Float16*)outv)[idx] = (_Float16)v;
                } else if (EPI == 1) {
                    ((float*)outv)[idx] = v * sc + resid[idx];
                } else {
                    float ge = 0.5f * v * (1.0f + erff(v * 0.70710678118654752f));
                    ((_Float16*)outv)[idx] = (_Float16)ge;
                }
            }
        }
}

// ---------------- flash attention (f16 in/out, f32 accum) ----------------
// grid: (B*H, S/64), block: 128 (4 waves), wave w owns 16 query rows.
__global__ __launch_bounds__(128) void attn_wmma(const _Float16* __restrict__ Q,
                                                 const _Float16* __restrict__ K,
                                                 const _Float16* __restrict__ V,
                                                 _Float16* __restrict__ O) {
    const int bh = blockIdx.x;
    const int b  = bh / Hc;
    const int h  = bh % Hc;
    const int q0 = blockIdx.y * 64;
    const int w    = threadIdx.x >> 5;
    const int lane = threadIdx.x & 31;
    const int lr   = lane & 15;
    const int hi   = lane >> 4;
    const int tid  = threadIdx.x;

    __shared__ float sP[4][16][68];                     // per-wave 16x64 score/prob tile
    __shared__ __align__(16) _Float16 Vt[64][72];       // V tile transposed: [d][t]
    __shared__ float cfac[4][16];
    __shared__ float linv[4][16];

    const size_t base = ((size_t)b * Sc) * Cc + (size_t)h * Dc;

    // Q A-fragments for both K-steps of D=64 (resident for whole kernel)
    Frag16 qa[2];
    {
        int qrow = q0 + w * 16 + lr;
        const _Float16* qp = Q + base + (size_t)qrow * Cc;
        for (int dc = 0; dc < 2; ++dc) {
            int d = dc * 32 + hi * 8;
            qa[dc].q[0] = *(const uint4*)(qp + d);
            qa[dc].q[1] = *(const uint4*)(qp + d + 16);
        }
    }

    v8f oacc[4];
    for (int dt = 0; dt < 4; ++dt)
        for (int r = 0; r < 8; ++r) oacc[dt][r] = 0.f;

    float mrow = -INFINITY;   // valid in lanes 0..15
    float lrow = 0.f;

    for (int t0 = 0; t0 < Sc; t0 += 64) {
        // ---- stage V tile (64 t x 64 d) transposed into LDS ----
        for (int itv = 0; itv < 4; ++itv) {
            int vi = itv * 128 + tid;        // 0..511 vectors of 8 halfs
            int t  = vi >> 3;
            int d8 = (vi & 7) * 8;
            H8 hv;
            hv.q = *(const uint4*)(V + base + (size_t)(t0 + t) * Cc + d8);
            for (int e = 0; e < 8; ++e)
                Vt[d8 + e][t] = hv.hh[e];
        }

        // ---- scores = Q K^T for 16 x 64 tile (hoisted loads, batched WMMA) ----
        Frag16 kf[2][4];
        for (int kc = 0; kc < 2; ++kc)
            for (int j = 0; j < 4; ++j) {
                int t = t0 + j * 16 + lr;
                const uint4* p = (const uint4*)(K + base + (size_t)t * Cc + kc * 32 + hi * 8);
                kf[kc][j].q[0] = p[0];
                kf[kc][j].q[1] = p[2];
            }
        v8f scr[4];
        for (int j = 0; j < 4; ++j)
            for (int r = 0; r < 8; ++r) scr[j][r] = 0.f;
        for (int kc = 0; kc < 2; ++kc)
            for (int j = 0; j < 4; ++j)
                scr[j] = wmma16(qa[kc].h, kf[kc][j].h, scr[j]);

        for (int j = 0; j < 4; ++j)
            for (int r = 0; r < 8; ++r)
                sP[w][r + hi * 8][j * 16 + lr] = scr[j][r] * 0.125f;   // 1/sqrt(64)
        __syncthreads();   // sP + Vt visible to all

        // ---- online softmax (one lane per query row) ----
        if (lane < 16) {
            float cmax = -INFINITY;
            for (int c = 0; c < 64; ++c) cmax = fmaxf(cmax, sP[w][lane][c]);
            float mnew = fmaxf(mrow, cmax);
            float corr = __expf(mrow - mnew);
            float ssum = 0.f;
            for (int c = 0; c < 64; ++c) {
                float p = __expf(sP[w][lane][c] - mnew);
                sP[w][lane][c] = p;
                ssum += p;
            }
            lrow = lrow * corr + ssum;
            mrow = mnew;
            cfac[w][lane] = corr;
        }
        __syncthreads();

        // rescale running output
        for (int r = 0; r < 8; ++r) {
            float cc = cfac[w][r + hi * 8];
            for (int dt = 0; dt < 4; ++dt) oacc[dt][r] *= cc;
        }

        // ---- O += P V (V fragments now contiguous 16B LDS reads) ----
        for (int ks = 0; ks < 2; ++ks) {
            Frag16 pa;
            for (int e = 0; e < 16; ++e) {
                int tl = ks * 32 + ((e < 8) ? 0 : 16) + hi * 8 + (e & 7);
                pa.h[e] = (_Float16)sP[w][lr][tl];
            }
            for (int dt = 0; dt < 4; ++dt) {
                Frag16 vb;
                int d = dt * 16 + lr;
                int tb = ks * 32 + hi * 8;
                vb.q[0] = *(const uint4*)(&Vt[d][tb]);
                vb.q[1] = *(const uint4*)(&Vt[d][tb + 16]);
                oacc[dt] = wmma16(pa.h, vb.h, oacc[dt]);
            }
        }
        __syncthreads();   // protect sP/Vt before next chunk overwrites
    }

    if (lane < 16) linv[w][lane] = 1.0f / lrow;
    __syncthreads();

    for (int dt = 0; dt < 4; ++dt)
        for (int r = 0; r < 8; ++r) {
            float v = oacc[dt][r] * linv[w][r + hi * 8];
            int m = q0 + w * 16 + r + hi * 8;
            O[base + (size_t)m * Cc + dt * 16 + lr] = (_Float16)v;
        }
}

// ---------------- host-side orchestration ----------------
extern "C" void kernel_launch(void* const* d_in, const int* in_sizes, int n_in,
                              void* d_out, int out_size, void* d_ws, size_t ws_size,
                              hipStream_t stream) {
    const float* hs    = (const float*)d_in[0];
    const float* ln1_w = (const float*)d_in[1];
    const float* ln1_b = (const float*)d_in[2];
    const float* q_w   = (const float*)d_in[3];
    const float* q_b   = (const float*)d_in[4];
    const float* k_w   = (const float*)d_in[5];
    const float* k_b   = (const float*)d_in[6];
    const float* v_w   = (const float*)d_in[7];
    const float* v_b   = (const float*)d_in[8];
    const float* o_w   = (const float*)d_in[9];
    const float* o_b   = (const float*)d_in[10];
    const float* lam1  = (const float*)d_in[11];
    const float* ln2_w = (const float*)d_in[12];
    const float* ln2_b = (const float*)d_in[13];
    const float* fc1_w = (const float*)d_in[14];
    const float* fc1_b = (const float*)d_in[15];
    const float* fc2_w = (const float*)d_in[16];
    const float* fc2_b = (const float*)d_in[17];
    const float* lam2  = (const float*)d_in[18];

    // carve workspace (256B aligned)
    char* base = (char*)d_ws;
    size_t off = 0;
    auto carve = [&](size_t bytes) -> char* {
        char* p = base + off;
        off = (off + bytes + 255) & ~(size_t)255;
        return p;
    };
    const size_t CCn = (size_t)Cc * Cc;        // 589824
    const size_t CFn = (size_t)Cc * Fc;        // 2359296
    const size_t MCn = (size_t)Mc * Cc;        // 6291456
    const size_t MFn = (size_t)Mc * Fc;        // 25165824

    _Float16* wq   = (_Float16*)carve(CCn * 2);
    _Float16* wk   = (_Float16*)carve(CCn * 2);
    _Float16* wv   = (_Float16*)carve(CCn * 2);
    _Float16* wo   = (_Float16*)carve(CCn * 2);
    _Float16* w1   = (_Float16*)carve(CFn * 2);
    _Float16* w2   = (_Float16*)carve(CFn * 2);
    _Float16* xb   = (_Float16*)carve(MCn * 2);   // LN1 out, reused for LN2 out
    _Float16* qb   = (_Float16*)carve(MCn * 2);
    _Float16* kb   = (_Float16*)carve(MCn * 2);
    _Float16* vb   = (_Float16*)carve(MCn * 2);
    _Float16* cb   = (_Float16*)carve(MCn * 2);   // attention context
    _Float16* act  = (_Float16*)carve(MFn * 2);   // gelu(fc1) out
    float*    hid  = (float*)   carve(MCn * 4);   // post-attention residual stream

    // 1) weights f32 -> f16
    cvt_f32_f16<<<(int)((CCn + 255) / 256), 256, 0, stream>>>(q_w,   wq, (int)CCn);
    cvt_f32_f16<<<(int)((CCn + 255) / 256), 256, 0, stream>>>(k_w,   wk, (int)CCn);
    cvt_f32_f16<<<(int)((CCn + 255) / 256), 256, 0, stream>>>(v_w,   wv, (int)CCn);
    cvt_f32_f16<<<(int)((CCn + 255) / 256), 256, 0, stream>>>(o_w,   wo, (int)CCn);
    cvt_f32_f16<<<(int)((CFn + 255) / 256), 256, 0, stream>>>(fc1_w, w1, (int)CFn);
    cvt_f32_f16<<<(int)((CFn + 255) / 256), 256, 0, stream>>>(fc2_w, w2, (int)CFn);

    // 2) LN1
    ln_f16<<<Mc, 256, 0, stream>>>(hs, ln1_w, ln1_b, xb);

    // 3) Q/K/V projections (f16 out, bias fused)
    dim3 gC(Cc / 128, Mc / 64);
    gemm_wmma_f16<0><<<gC, 256, 0, stream>>>(xb, wq, q_b, nullptr, nullptr, qb, Cc, Cc);
    gemm_wmma_f16<0><<<gC, 256, 0, stream>>>(xb, wk, k_b, nullptr, nullptr, kb, Cc, Cc);
    gemm_wmma_f16<0><<<gC, 256, 0, stream>>>(xb, wv, v_b, nullptr, nullptr, vb, Cc, Cc);

    // 4) attention
    attn_wmma<<<dim3(Bc * Hc, Sc / 64), 128, 0, stream>>>(qb, kb, vb, cb);

    // 5) O projection + layer_scale1 + residual -> hidden (f32)
    gemm_wmma_f16<1><<<gC, 256, 0, stream>>>(cb, wo, o_b, lam1, hs, hid, Cc, Cc);

    // 6) LN2
    ln_f16<<<Mc, 256, 0, stream>>>(hid, ln2_w, ln2_b, xb);

    // 7) FC1 + exact GELU (f16 out)
    dim3 gF(Fc / 128, Mc / 64);
    gemm_wmma_f16<2><<<gF, 256, 0, stream>>>(xb, w1, fc1_b, nullptr, nullptr, act, Fc, Cc);

    // 8) FC2 + layer_scale2 + residual -> d_out (f32)
    gemm_wmma_f16<1><<<gC, 256, 0, stream>>>(act, w2, fc2_b, lam2, hid, (float*)d_out, Cc, Fc);
}
